// BasicBlock_69647189672216
// MI455X (gfx1250) — compile-verified, matrix operand order checked by reading, not run on hardware
//
#include <hip/hip_runtime.h>
#include <hip/hip_bf16.h>

typedef __attribute__((ext_vector_type(16))) __bf16 v16bf;
typedef __attribute__((ext_vector_type(8)))  __bf16 v8bf;
typedef __attribute__((ext_vector_type(8)))  float  v8f;

#define NB    1700      // nodes
#define BB    64        // batch
#define CC    256       // channels
#define MM    (BB*NB)   // 108800 GEMM rows (multiple of 128: 850*128)

// ---------------------------------------------------------------------------
// Convert W [256x256] f32 row-major (K x N) -> bf16 transposed (N x K)
// so that B-tiles load with the same contiguous-chunk pattern as A-tiles.
// ---------------------------------------------------------------------------
__global__ void k_wcvt(const float* __restrict__ W, __bf16* __restrict__ WT) {
    int idx = blockIdx.x * 256 + threadIdx.x;     // 65536 elements
    int n = idx >> 8;
    int k = idx & 255;
    WT[n * 256 + k] = (__bf16)W[k * 256 + n];
}

// ---------------------------------------------------------------------------
// CSR build over destination (col) nodes. deg = in_count + 1 (self loop).
// Deterministic: each node scans edges in order (no atomics anywhere).
// ---------------------------------------------------------------------------
__global__ void k_count(const int* __restrict__ col, int E,
                        int* __restrict__ off, float* __restrict__ dinv) {
    int n = blockIdx.x * 256 + threadIdx.x;
    if (n >= NB) return;
    int cnt = 0;
    for (int e = 0; e < E; ++e) cnt += (col[e] == n);
    off[n + 1] = cnt;
    dinv[n] = rsqrtf((float)(cnt + 1));           // +1 self loop, always > 0
}

__global__ void k_scan(int* __restrict__ off) {
    if (threadIdx.x != 0 || blockIdx.x != 0) return;
    int run = 0;
    off[0] = 0;
    for (int n = 0; n < NB; ++n) { run += off[n + 1]; off[n + 1] = run; }
}

__global__ void k_fill(const int* __restrict__ row, const int* __restrict__ col, int E,
                       const int* __restrict__ off, const float* __restrict__ dinv,
                       int* __restrict__ csrc, float* __restrict__ cw) {
    int n = blockIdx.x * 256 + threadIdx.x;
    if (n >= NB) return;
    int p  = off[n];
    float dn = dinv[n];
    for (int e = 0; e < E; ++e) {
        if (col[e] == n) {
            int r = row[e];
            csrc[p] = r;
            cw[p]   = dinv[r] * dn;
            p++;
        }
    }
}

// ---------------------------------------------------------------------------
// GEMM: H[M x 256] (bf16 out) = act(A[M x 256] f32) @ WT^T  via bf16 WMMA.
// Block 256 thr = 8 waves; block tile 128x128; wave tile 32x64; K-step 32.
// ---------------------------------------------------------------------------
__global__ __launch_bounds__(256)
void k_gemm(const float* __restrict__ A, const __bf16* __restrict__ WT,
            __bf16* __restrict__ H, int applyRelu) {
    __shared__ __bf16 As[128][40];   // padded stride: 80B (20 dwords) -> no bank conflicts
    __shared__ __bf16 Bs[128][40];

    const int tid   = threadIdx.x;
    const int wave  = tid >> 5;
    const int lane  = tid & 31;
    const int waveM = wave & 3;      // 4 waves along M (32 rows each)
    const int waveN = wave >> 2;     // 2 waves along N (64 cols each)
    const int m0    = blockIdx.y * 128;
    const int n0    = blockIdx.x * 128;

    const int half = lane >> 4;      // chunk base selector per ISA A-layout
    const int lr   = lane & 15;
    const int kb   = half * 8;

    v8f acc[2][4];
    #pragma unroll
    for (int m = 0; m < 2; ++m)
        #pragma unroll
        for (int n = 0; n < 4; ++n)
            acc[m][n] = (v8f)(0.0f);

    for (int k0 = 0; k0 < 256; k0 += 32) {
        // ---- A tile: 128x32 f32 -> bf16 (optional ReLU) ----
        #pragma unroll
        for (int i = 0; i < 4; ++i) {
            int idx = tid + i * 256;           // 1024 float4 chunks
            int r   = idx >> 3;
            int c4  = (idx & 7) << 2;
            float4 v = *(const float4*)&A[(size_t)(m0 + r) * 256 + k0 + c4];
            if (k0 < 224)  // prefetch next K-slice of A
                __builtin_prefetch(&A[(size_t)(m0 + r) * 256 + k0 + 32 + c4], 0, 0);
            if (applyRelu) {
                v.x = fmaxf(v.x, 0.f); v.y = fmaxf(v.y, 0.f);
                v.z = fmaxf(v.z, 0.f); v.w = fmaxf(v.w, 0.f);
            }
            As[r][c4 + 0] = (__bf16)v.x; As[r][c4 + 1] = (__bf16)v.y;
            As[r][c4 + 2] = (__bf16)v.z; As[r][c4 + 3] = (__bf16)v.w;
        }
        // ---- B tile: 128(N) x 32(K) bf16, contiguous 16B chunks ----
        #pragma unroll
        for (int i = 0; i < 2; ++i) {
            int idx = tid + i * 256;           // 512 chunks of 8 bf16
            int r   = idx >> 2;
            int c8  = (idx & 3) << 3;
            uint4 v = *(const uint4*)&WT[(size_t)(n0 + r) * 256 + k0 + c8];
            *(uint4*)&Bs[r][c8] = v;
        }
        __syncthreads();

        v16bf afrag[2], bfrag[4];
        #pragma unroll
        for (int m = 0; m < 2; ++m) {
            int r = waveM * 32 + m * 16 + lr;
            v8bf lo = *(const v8bf*)&As[r][kb];
            v8bf hi = *(const v8bf*)&As[r][kb + 16];
            afrag[m] = __builtin_shufflevector(lo, hi, 0,1,2,3,4,5,6,7,8,9,10,11,12,13,14,15);
        }
        #pragma unroll
        for (int n = 0; n < 4; ++n) {
            int r = waveN * 64 + n * 16 + lr;
            v8bf lo = *(const v8bf*)&Bs[r][kb];
            v8bf hi = *(const v8bf*)&Bs[r][kb + 16];
            bfrag[n] = __builtin_shufflevector(lo, hi, 0,1,2,3,4,5,6,7,8,9,10,11,12,13,14,15);
        }
        #pragma unroll
        for (int m = 0; m < 2; ++m)
            #pragma unroll
            for (int n = 0; n < 4; ++n)
                acc[m][n] = __builtin_amdgcn_wmma_f32_16x16x32_bf16(
                    false, afrag[m], false, bfrag[n],
                    (short)0, acc[m][n], false, false);
        __syncthreads();
    }

    // ---- epilogue: f32 acc -> bf16 H ----
    #pragma unroll
    for (int m = 0; m < 2; ++m)
        #pragma unroll
        for (int n = 0; n < 4; ++n) {
            int col = n0 + waveN * 64 + n * 16 + lr;
            #pragma unroll
            for (int i = 0; i < 8; ++i) {
                int rw = m0 + waveM * 32 + m * 16 + i + half * 8;
                H[(size_t)rw * 256 + col] = (__bf16)acc[m][n][i];
            }
        }
}

// ---------------------------------------------------------------------------
// Gather-reduce: block = (node, 8-batch group); lane owns 8 contiguous
// channels (one 16B b128 load per edge). 32 lanes x 8 ch = 256 channels;
// 8 waves = 8 batches. Prefetch next edge's source row (global_prefetch).
// out[b,n,c] = bias[c] + dinv[n]^2*h[b,n,c] + sum_j w[j]*h[b,src[j],c]
// If residual != null: out = relu(out + residual) (final layer).
// ---------------------------------------------------------------------------
__global__ __launch_bounds__(256)
void k_gather(const __bf16* __restrict__ h,
              const int* __restrict__ off, const int* __restrict__ csrc,
              const float* __restrict__ cw, const float* __restrict__ dinv,
              const float* __restrict__ bias, const float* __restrict__ residual,
              float* __restrict__ out) {
    const int n    = blockIdx.x;
    const int b    = blockIdx.y * 8 + (threadIdx.x >> 5);
    const int c8   = (threadIdx.x & 31) << 3;          // 8 channels per lane
    const size_t rowbase = ((size_t)b * NB) * CC;

    float acc[8];
    {
        float4 b0 = *(const float4*)&bias[c8];
        float4 b1 = *(const float4*)&bias[c8 + 4];
        acc[0] = b0.x; acc[1] = b0.y; acc[2] = b0.z; acc[3] = b0.w;
        acc[4] = b1.x; acc[5] = b1.y; acc[6] = b1.z; acc[7] = b1.w;
    }

    // self loop: dinv[n]^2 * h[b,n,:]
    const float dv  = dinv[n];
    const float dv2 = dv * dv;
    {
        v8bf p = *(const v8bf*)&h[rowbase + (size_t)n * CC + c8];
        #pragma unroll
        for (int i = 0; i < 8; ++i) acc[i] += dv2 * (float)p[i];
    }

    const int j0 = off[n], j1 = off[n + 1];
    for (int j = j0; j < j1; ++j) {
        if (j + 1 < j1)
            __builtin_prefetch(&h[rowbase + (size_t)csrc[j + 1] * CC + c8], 0, 0);
        const int   src = csrc[j];
        const float wgt = cw[j];
        v8bf p = *(const v8bf*)&h[rowbase + (size_t)src * CC + c8];
        #pragma unroll
        for (int i = 0; i < 8; ++i) acc[i] += wgt * (float)p[i];
    }

    const size_t oidx = rowbase + (size_t)n * CC + c8;
    if (residual) {
        float4 r0 = *(const float4*)&residual[oidx];
        float4 r1 = *(const float4*)&residual[oidx + 4];
        acc[0] = fmaxf(acc[0] + r0.x, 0.f); acc[1] = fmaxf(acc[1] + r0.y, 0.f);
        acc[2] = fmaxf(acc[2] + r0.z, 0.f); acc[3] = fmaxf(acc[3] + r0.w, 0.f);
        acc[4] = fmaxf(acc[4] + r1.x, 0.f); acc[5] = fmaxf(acc[5] + r1.y, 0.f);
        acc[6] = fmaxf(acc[6] + r1.z, 0.f); acc[7] = fmaxf(acc[7] + r1.w, 0.f);
    }
    float4 o0 = make_float4(acc[0], acc[1], acc[2], acc[3]);
    float4 o1 = make_float4(acc[4], acc[5], acc[6], acc[7]);
    *(float4*)&out[oidx]     = o0;
    *(float4*)&out[oidx + 4] = o1;
}

// ---------------------------------------------------------------------------
extern "C" void kernel_launch(void* const* d_in, const int* in_sizes, int n_in,
                              void* d_out, int out_size, void* d_ws, size_t ws_size,
                              hipStream_t stream) {
    const int E_SP = 6400, E_TM = 10000;

    const float* x  = (const float*)d_in[0];
    const float* W1 = (const float*)d_in[1];
    const float* b1 = (const float*)d_in[2];
    const float* W2 = (const float*)d_in[3];
    const float* b2 = (const float*)d_in[4];
    const float* W3 = (const float*)d_in[5];
    const float* b3 = (const float*)d_in[6];
    const int* esp  = (const int*)d_in[7];   // [2, 6400]
    const int* etm  = (const int*)d_in[8];   // [2, 10000]
    float* outp     = (float*)d_out;

    // workspace carve-up (256B aligned)
    char* w = (char*)d_ws;
    auto alloc = [&](size_t bytes) -> char* {
        char* p = w;
        w += (bytes + 255) & ~(size_t)255;
        return p;
    };
    __bf16* h    = (__bf16*)alloc((size_t)MM * CC * 2);   // 55.7 MB
    float*  s    = (float*) alloc((size_t)MM * CC * 4);   // 111.4 MB
    __bf16* WT1  = (__bf16*)alloc(65536 * 2);
    __bf16* WT2  = (__bf16*)alloc(65536 * 2);
    __bf16* WT3  = (__bf16*)alloc(65536 * 2);
    int*    off_sp  = (int*)  alloc((NB + 1) * 4);
    float*  dinv_sp = (float*)alloc(NB * 4);
    int*    src_sp  = (int*)  alloc(E_SP * 4);
    float*  cw_sp   = (float*)alloc(E_SP * 4);
    int*    off_tm  = (int*)  alloc((NB + 1) * 4);
    float*  dinv_tm = (float*)alloc(NB * 4);
    int*    src_tm  = (int*)  alloc(E_TM * 4);
    float*  cw_tm   = (float*)alloc(E_TM * 4);

    // weights -> bf16 transposed
    k_wcvt<<<256, 256, 0, stream>>>(W1, WT1);
    k_wcvt<<<256, 256, 0, stream>>>(W2, WT2);
    k_wcvt<<<256, 256, 0, stream>>>(W3, WT3);

    // CSR for both edge sets (row = e[0..E), col = e[E..2E))
    dim3 gN((NB + 255) / 256);
    k_count<<<gN, 256, 0, stream>>>(esp + E_SP, E_SP, off_sp, dinv_sp);
    k_scan <<<1, 1, 0, stream>>>(off_sp);
    k_fill <<<gN, 256, 0, stream>>>(esp, esp + E_SP, E_SP, off_sp, dinv_sp, src_sp, cw_sp);

    k_count<<<gN, 256, 0, stream>>>(etm + E_TM, E_TM, off_tm, dinv_tm);
    k_scan <<<1, 1, 0, stream>>>(off_tm);
    k_fill <<<gN, 256, 0, stream>>>(etm, etm + E_TM, E_TM, off_tm, dinv_tm, src_tm, cw_tm);

    dim3 gGemm(2, MM / 128);      // N tiles x M tiles (108800 = 850*128)
    dim3 gGath(NB, BB / 8);       // (1700, 8): 8 batches per block

    // layer 1: h = x @ W1 ; s = Agg_sp(h) + b1          (ReLU folded into next GEMM)
    k_gemm  <<<gGemm, 256, 0, stream>>>(x, WT1, h, 0);
    k_gather<<<gGath, 256, 0, stream>>>(h, off_sp, src_sp, cw_sp, dinv_sp, b1, nullptr, s);

    // layer 2: h = relu(s) @ W2 ; s = Agg_tm(h) + b2
    k_gemm  <<<gGemm, 256, 0, stream>>>(s, WT2, h, 1);
    k_gather<<<gGath, 256, 0, stream>>>(h, off_tm, src_tm, cw_tm, dinv_tm, b2, nullptr, s);

    // layer 3: h = relu(s) @ W3 ; out = relu(Agg_sp(h) + b3 + x)
    k_gemm  <<<gGemm, 256, 0, stream>>>(s, WT3, h, 1);
    k_gather<<<gGath, 256, 0, stream>>>(h, off_sp, src_sp, cw_sp, dinv_sp, b3, x, outp);
}
